// DenseSAKELayer_20624432955637
// MI455X (gfx1250) — matrix operand order repaired and checked by reading, not compiled, not run on hardware
//
#include <hip/hip_runtime.h>
#include <hip/hip_bf16.h>

typedef __attribute__((ext_vector_type(16))) _Float16 v16h;
typedef __attribute__((ext_vector_type(8)))  _Float16 v8h;
typedef __attribute__((ext_vector_type(4)))  _Float16 v4h;
typedef __attribute__((ext_vector_type(8)))  float    v8f;

#define INFV 100000.0f

// fast reciprocal: single v_rcp_f32 (~1 ulp), avoids IEEE div expansion
__device__ __forceinline__ float fast_rcp(float x) { return __builtin_amdgcn_rcpf(x); }
__device__ __forceinline__ float silu_f(float x) {
    return x * fast_rcp(1.0f + __expf(-x));
}

// ---- WMMA helpers (CDNA5 16x16x32 f16, f32 accumulate) -------------------
__device__ __forceinline__ v8f wmma_f16(v16h a, v16h b, v8f c) {
    return __builtin_amdgcn_wmma_f32_16x16x32_f16(
        /*neg_a=*/false, a, /*neg_b=*/false, b,
        /*c_mod=*/(short)0, c, /*reuse_a=*/false, /*reuse_b=*/false);
}

// A-matrix 16x32 f16 fragment from LDS row-major [M][lda] (contiguous -> b128).
// lanes 0-15 -> M=lane, K = {k0..k0+15}
// lanes 16-31 -> M=lane-16, K = {k0+16..k0+31}
__device__ __forceinline__ v16h load_a16(const _Float16* sA, int lda, int m0, int k0, int lane) {
    int grp = lane >> 4;
    const _Float16* row = sA + (m0 + (lane & 15)) * lda;
    v16h a;
    int kb0 = k0 + (grp ? 16 : 0);
#pragma unroll
    for (int t = 0; t < 8; ++t) a[t] = row[kb0 + t];
    int kb1 = k0 + 8 + (grp ? 16 : 0);
#pragma unroll
    for (int t = 0; t < 8; ++t) a[8 + t] = row[kb1 + t];
    return a;
}

// B-matrix 32x16 fragment from LDS stored TRANSPOSED [N][ldk] f16.
// lane needs N = n0+(lane&15), K = kb..kb+15 -> 16 contiguous halves (2x b128).
__device__ __forceinline__ v16h load_bT16(const _Float16* sBT, int ldk, int k0, int n0, int lane) {
    int grp = lane >> 4;
    const _Float16* row = sBT + (n0 + (lane & 15)) * ldk;
    int kb = k0 + (grp ? 16 : 0);
    v16h b;
#pragma unroll
    for (int t = 0; t < 16; ++t) b[t] = row[kb + t];
    return b;
}

// Block-wide reduction of 4 floats across 256 threads. op: 0=max, 1=sum.
__device__ __forceinline__ void block_reduce4(float* v, float* sRed, int op) {
    int t = threadIdx.x;
    sRed[t * 4 + 0] = v[0]; sRed[t * 4 + 1] = v[1];
    sRed[t * 4 + 2] = v[2]; sRed[t * 4 + 3] = v[3];
    __syncthreads();
    for (int s = 128; s > 0; s >>= 1) {
        if (t < s) {
#pragma unroll
            for (int q = 0; q < 4; ++q) {
                float a = sRed[t * 4 + q], b = sRed[(t + s) * 4 + q];
                sRed[t * 4 + q] = op ? (a + b) : fmaxf(a, b);
            }
        }
        __syncthreads();
    }
    v[0] = sRed[0]; v[1] = sRed[1]; v[2] = sRed[2]; v[3] = sRed[3];
    __syncthreads();
}

// ===========================================================================
// Kernel 1: per-node precompute: lin_j, lin_i (h @ W_e1 halves), velocity gate
// ===========================================================================
extern "C" __global__ __launch_bounds__(256)
void sake_pre(const float* __restrict__ h, const float* __restrict__ W_e1,
              const float* __restrict__ W_v1, const float* __restrict__ b_v1,
              const float* __restrict__ W_v2,
              float* __restrict__ linj, float* __restrict__ lini,
              float* __restrict__ vgate) {
    __shared__ float sH[64];
    __shared__ float sT[64];
    int node = blockIdx.x, tid = threadIdx.x;
    if (tid < 64) sH[tid] = h[node * 64 + tid];
    __syncthreads();
    if (tid < 128) {
        int k = tid & 63;
        bool second = tid >= 64;
        const float* W = W_e1 + (second ? 64 * 64 : 0);
        float acc = 0.0f;
#pragma unroll 8
        for (int f = 0; f < 64; ++f) acc += sH[f] * W[f * 64 + k];
        (second ? lini : linj)[node * 64 + k] = acc;
    } else if (tid < 192) {
        int hh = tid - 128;
        float acc = b_v1[hh];
#pragma unroll 8
        for (int f = 0; f < 64; ++f) acc += sH[f] * W_v1[f * 64 + hh];
        sT[hh] = silu_f(acc) * W_v2[hh];
    }
    __syncthreads();
    if (tid == 0) {
        float g = 0.0f;
        for (int hh = 0; hh < 64; ++hh) g += sT[hh];
        vgate[node] = g;
    }
}

// ===========================================================================
// Kernel 2: fused per-(b,i) edge panel. One workgroup per node (1024 WGs).
// Dynamic LDS 135168 B (see offsets below).
// ===========================================================================
extern "C" __global__ __launch_bounds__(256)
void sake_edge(const float* __restrict__ x,
               const float* __restrict__ linj, const float* __restrict__ lini,
               const float* __restrict__ W_e1, const float* __restrict__ b_e1,
               const float* __restrict__ W_e2, const float* __restrict__ b_e2,
               const float* __restrict__ W_c1, const float* __restrict__ b_c1,
               const float* __restrict__ W_c2, const float* __restrict__ b_c2,
               const float* __restrict__ W_s,  const float* __restrict__ b_s,
               const float* __restrict__ log_gamma,
               float* __restrict__ hE, float* __restrict__ combN,
               float* __restrict__ dV) {
    extern __shared__ char smem[];
    _Float16* sA    = (_Float16*)(smem);              // [256][64]  32KB
    _Float16* sHe   = (_Float16*)(smem + 32768);      // [256][64]  32KB
    _Float16* sWe2T = (_Float16*)(smem + 65536);      // [64][64]   8KB (W_e2^T)
    _Float16* sWc1T = (_Float16*)(smem + 73728);      // [64][64]   8KB (W_c1^T)
    _Float16* sWc2T = (_Float16*)(smem + 81920);      // [32][64]   4KB (W_c2^T)
    float*    sCo   = (float*)(smem + 86016);         // [256][32]  32KB
    float*    sAtt  = (float*)(smem + 118784);        // [256][4]   4KB
    float*    sRed  = (float*)(smem + 122880);        // [1024]     4KB (reduce+partials)
    float*    sNrm  = (float*)(smem + 126976);        // [256]      1KB
    float*    sXs   = (float*)(smem + 128000);        // [256][3]   3KB
    float*    sM    = (float*)(smem + 131072);        // misc       4KB
    float2* sPW  = (float2*)sM;      // [64] {lin_i+b_e1, w_row}
    float*  sBe2 = sM + 128;         // 64
    float*  sBc1 = sM + 192;         // 64
    float*  sBc2 = sM + 256;         // 32
    float*  sWs  = sM + 288;         // 256 ([64][4])
    float*  sBs  = sM + 544;         // 4
    float*  sGam = sM + 548;         // 4
    float*  sCS  = sM + 552;         // 96 ([3][32])

    const int tid  = threadIdx.x;
    const int node = blockIdx.x;          // b*256 + i
    const int b    = node >> 8;
    const int i    = node & 255;
    const int wave = tid >> 5, lane = tid & 31, grp = lane >> 4;

    // ---- stage weights transposed (f32 -> f16), coalesced global reads --
    for (int e = tid; e < 4096; e += 256) {
        int k = e >> 6, n = e & 63;
        sWe2T[n * 64 + k] = (_Float16)W_e2[e];
        sWc1T[n * 64 + k] = (_Float16)W_c1[e];
    }
    for (int e = tid; e < 2048; e += 256) {
        int k = e >> 5, n = e & 31;
        sWc2T[n * 64 + k] = (_Float16)W_c2[e];
    }
    if (tid < 64) {
        float2 pw;
        pw.x = lini[node * 64 + tid] + b_e1[tid];
        pw.y = W_e1[128 * 64 + tid];
        sPW[tid]  = pw;
        sBe2[tid] = b_e2[tid];
        sBc1[tid] = b_c1[tid];
    }
    if (tid < 32) sBc2[tid] = b_c2[tid];
    if (tid < 256) sWs[tid] = W_s[tid];
    if (tid < 4) { sBs[tid] = b_s[tid]; sGam[tid] = __expf(log_gamma[tid]); }

    // ---- geometry: thread j owns row j ----------------------------------
    {
        int j = tid;
        float xi0 = x[(b * 256 + i) * 3 + 0];
        float xi1 = x[(b * 256 + i) * 3 + 1];
        float xi2 = x[(b * 256 + i) * 3 + 2];
        float d0 = x[(b * 256 + j) * 3 + 0] - xi0;
        float d1 = x[(b * 256 + j) * 3 + 1] - xi1;
        float d2 = x[(b * 256 + j) * 3 + 2] - xi2;
        float nr = sqrtf(d0 * d0 + d1 * d1 + d2 * d2 + 1e-5f);
        sNrm[j] = nr;
        float den = (nr + 1e-5f);
        float inv = fast_rcp(den * den);
        sXs[j * 3 + 0] = d0 * inv;
        sXs[j * 3 + 1] = d1 * inv;
        sXs[j * 3 + 2] = d2 * inv;
    }
    __syncthreads();

    // ---- A = silu(pre) f16; per-thread row-contiguous, b128 LDS stores --
    {
        int j = tid;
        float nr = sNrm[j];
        const float4* lp = (const float4*)(linj + (size_t)(b * 256 + j) * 64);
        v8h* dst = (v8h*)(sA + j * 64);
#pragma unroll
        for (int c8 = 0; c8 < 8; ++c8) {
            float4 f0 = lp[c8 * 2 + 0];
            float4 f1 = lp[c8 * 2 + 1];
            float in[8] = {f0.x, f0.y, f0.z, f0.w, f1.x, f1.y, f1.z, f1.w};
            v8h o;
#pragma unroll
            for (int r = 0; r < 8; ++r) {
                float2 pw = sPW[c8 * 8 + r];
                o[r] = (_Float16)silu_f(in[r] + pw.x + nr * pw.y);
            }
            dst[c8] = o;
        }
    }
    __syncthreads();

    // ---- GEMM1: He = silu(A[256x64] @ We2[64x64] + be2) -----------------
#pragma unroll
    for (int mi = 0; mi < 2; ++mi) {
        int m0 = (wave * 2 + mi) * 16;
        v16h a0 = load_a16(sA, 64, m0, 0, lane);
        v16h a1 = load_a16(sA, 64, m0, 32, lane);
#pragma unroll
        for (int nt = 0; nt < 4; ++nt) {
            v8f c = 0.0f;
            c = wmma_f16(a0, load_bT16(sWe2T, 64, 0, nt * 16, lane), c);
            c = wmma_f16(a1, load_bT16(sWe2T, 64, 32, nt * 16, lane), c);
            int n = nt * 16 + (lane & 15);
#pragma unroll
            for (int r = 0; r < 8; ++r) {
                int m = m0 + r + (grp << 3);
                sHe[m * 64 + n] = (_Float16)silu_f(c[r] + sBe2[n]);
            }
        }
    }
    __syncthreads();

    // ---- semantic logits s[j][nh] (thread j; b128 row loads) ------------
    float s4[4] = {sBs[0], sBs[1], sBs[2], sBs[3]};
    {
        const v8h* rowp = (const v8h*)(sHe + tid * 64);
        const float4* wsp = (const float4*)sWs;
#pragma unroll
        for (int c8 = 0; c8 < 8; ++c8) {
            v8h hv = rowp[c8];
#pragma unroll
            for (int r = 0; r < 8; ++r) {
                float he = (float)hv[r];
                float4 w = wsp[c8 * 8 + r];
                s4[0] += he * w.x; s4[1] += he * w.y;
                s4[2] += he * w.z; s4[3] += he * w.w;
            }
        }
    }

    // ---- GEMM2: T = silu(He @ Wc1 + bc1) -> sA --------------------------
#pragma unroll
    for (int mi = 0; mi < 2; ++mi) {
        int m0 = (wave * 2 + mi) * 16;
        v16h a0 = load_a16(sHe, 64, m0, 0, lane);
        v16h a1 = load_a16(sHe, 64, m0, 32, lane);
#pragma unroll
        for (int nt = 0; nt < 4; ++nt) {
            v8f c = 0.0f;
            c = wmma_f16(a0, load_bT16(sWc1T, 64, 0, nt * 16, lane), c);
            c = wmma_f16(a1, load_bT16(sWc1T, 64, 32, nt * 16, lane), c);
            int n = nt * 16 + (lane & 15);
#pragma unroll
            for (int r = 0; r < 8; ++r) {
                int m = m0 + r + (grp << 3);
                sA[m * 64 + n] = (_Float16)silu_f(c[r] + sBc1[n]);
            }
        }
    }
    __syncthreads();

    // ---- GEMM3: coeff = T @ Wc2 + bc2 -> sCo f32 [256][32] --------------
#pragma unroll
    for (int mi = 0; mi < 2; ++mi) {
        int m0 = (wave * 2 + mi) * 16;
        v16h a0 = load_a16(sA, 64, m0, 0, lane);
        v16h a1 = load_a16(sA, 64, m0, 32, lane);
#pragma unroll
        for (int nt = 0; nt < 2; ++nt) {
            v8f c = 0.0f;
            c = wmma_f16(a0, load_bT16(sWc2T, 64, 0, nt * 16, lane), c);
            c = wmma_f16(a1, load_bT16(sWc2T, 64, 32, nt * 16, lane), c);
            int n = nt * 16 + (lane & 15);
#pragma unroll
            for (int r = 0; r < 8; ++r) {
                int m = m0 + r + (grp << 3);
                sCo[m * 32 + n] = c[r] + sBc2[n];
            }
        }
    }
    __syncthreads();

    // ---- attention chain (thread j owns row j; block reductions) --------
    {
        float nr = sNrm[tid];
        float diag = (tid == i) ? INFV : 0.0f;
        float le[4], ae[4], as_[4], at[4], tmp[4];
#pragma unroll
        for (int q = 0; q < 4; ++q) le[q] = -(nr + diag) * sGam[q];
#pragma unroll
        for (int q = 0; q < 4; ++q) tmp[q] = le[q];
        block_reduce4(tmp, sRed, 0);
#pragma unroll
        for (int q = 0; q < 4; ++q) ae[q] = __expf(le[q] - tmp[q]);
#pragma unroll
        for (int q = 0; q < 4; ++q) tmp[q] = ae[q];
        block_reduce4(tmp, sRed, 1);
#pragma unroll
        for (int q = 0; q < 4; ++q) ae[q] *= fast_rcp(tmp[q]);

        float ls[4];
#pragma unroll
        for (int q = 0; q < 4; ++q) {
            float z = s4[q] > 0.0f ? s4[q] : 0.2f * s4[q];   // LeakyReLU(0.2)
            ls[q] = z - diag;
        }
#pragma unroll
        for (int q = 0; q < 4; ++q) tmp[q] = ls[q];
        block_reduce4(tmp, sRed, 0);
#pragma unroll
        for (int q = 0; q < 4; ++q) as_[q] = __expf(ls[q] - tmp[q]);
#pragma unroll
        for (int q = 0; q < 4; ++q) tmp[q] = as_[q];
        block_reduce4(tmp, sRed, 1);
#pragma unroll
        for (int q = 0; q < 4; ++q) as_[q] *= fast_rcp(tmp[q]);

        float al[4];
#pragma unroll
        for (int q = 0; q < 4; ++q) al[q] = ae[q] * as_[q];
#pragma unroll
        for (int q = 0; q < 4; ++q) tmp[q] = al[q];
        block_reduce4(tmp, sRed, 0);
#pragma unroll
        for (int q = 0; q < 4; ++q) at[q] = __expf(al[q] - tmp[q]);
#pragma unroll
        for (int q = 0; q < 4; ++q) tmp[q] = at[q];
        block_reduce4(tmp, sRed, 1);
#pragma unroll
        for (int q = 0; q < 4; ++q) sAtt[tid * 4 + q] = at[q] * fast_rcp(tmp[q]);
    }
    __syncthreads();

    // ---- h_e aggregation: h_e[hh*4+nh] = sum_j He[j][hh]*att[j][nh] -----
    // thread: 4 consecutive hh (b64 loads), one nh, one j-quarter; LDS partials
    {
        int q  = tid & 15;          // hh block -> hh = 4q..4q+3
        int nh = (tid >> 4) & 3;
        int jc = tid >> 6;          // j-quarter
        float acc[4] = {0.0f, 0.0f, 0.0f, 0.0f};
        for (int j = jc * 64; j < jc * 64 + 64; ++j) {
            v4h hv = *(const v4h*)(sHe + j * 64 + q * 4);
            float a = sAtt[j * 4 + nh];
#pragma unroll
            for (int r = 0; r < 4; ++r) acc[r] += (float)hv[r] * a;
        }
#pragma unroll
        for (int r = 0; r < 4; ++r)
            sRed[(jc * 4 + nh) * 64 + q * 4 + r] = acc[r];
    }
    __syncthreads();
    {
        int hh = tid >> 2, nh = tid & 3;
        float acc = 0.0f;
#pragma unroll
        for (int jc = 0; jc < 4; ++jc)
            acc += sRed[(jc * 4 + nh) * 64 + hh];
        hE[node * 256 + tid] = acc;   // tid == hh*4+nh
    }
    __syncthreads();

    // ---- coefficient * geometry reductions (j-chunked partials) ---------
    {
        int c  = tid >> 3;          // 0..31
        int jc = tid & 7;           // j-eighth
        float a0 = 0.0f, a1 = 0.0f, a2 = 0.0f;
        for (int j = jc * 32; j < jc * 32 + 32; ++j) {
            float co = sCo[j * 32 + c];
            a0 += co * sXs[j * 3 + 0];
            a1 += co * sXs[j * 3 + 1];
            a2 += co * sXs[j * 3 + 2];
        }
        int p = (c * 8 + jc) * 3;
        sRed[p + 0] = a0; sRed[p + 1] = a1; sRed[p + 2] = a2;
    }
    __syncthreads();
    if (tid < 32) {
        float cs0 = 0.0f, cs1 = 0.0f, cs2 = 0.0f;
#pragma unroll
        for (int jc = 0; jc < 8; ++jc) {
            int p = (tid * 8 + jc) * 3;
            cs0 += sRed[p + 0]; cs1 += sRed[p + 1]; cs2 += sRed[p + 2];
        }
        combN[node * 32 + tid] = cs0 * cs0 + cs1 * cs1 + cs2 * cs2;
        sCS[0 * 32 + tid] = cs0; sCS[1 * 32 + tid] = cs1; sCS[2 * 32 + tid] = cs2;
    }
    __syncthreads();
    if (tid < 3) {
        float acc = 0.0f;
        for (int c = 0; c < 32; ++c) acc += sCS[tid * 32 + c];
        dV[node * 3 + tid] = acc * (1.0f / (256.0f * 32.0f));
    }
}

// ===========================================================================
// Kernel 3: node update. 16 WGs x 64 nodes. WMMA node-MLP (K=384), residual,
// post-norm MLP (scalar), velocity / coordinate update. Dynamic LDS 139776 B.
// ===========================================================================
extern "C" __global__ __launch_bounds__(256)
void sake_node(const float* __restrict__ h, const float* __restrict__ x,
               const float* __restrict__ v,
               const float* __restrict__ W_p1, const float* __restrict__ b_p1,
               const float* __restrict__ W_p2, const float* __restrict__ b_p2,
               const float* __restrict__ W_n1, const float* __restrict__ b_n1,
               const float* __restrict__ W_n2, const float* __restrict__ b_n2,
               const float* __restrict__ hE, const float* __restrict__ combN,
               const float* __restrict__ dV, const float* __restrict__ vgate,
               float* __restrict__ out_h, float* __restrict__ out_x,
               float* __restrict__ out_v) {
    extern __shared__ char smem[];
    _Float16* sIn  = (_Float16*)(smem);               // [64][384] 48KB
    _Float16* sW1T = (_Float16*)(smem + 49152);       // [64][384] 48KB (W_n1^T)
    _Float16* sT   = (_Float16*)(smem + 98304);       // [64][64]  8KB
    _Float16* sW2T = (_Float16*)(smem + 106496);      // [64][64]  8KB (W_n2^T)
    float*    sHC  = (float*)(smem + 114688);         // [64][64]  16KB
    float*    sCN  = (float*)(smem + 131072);         // [64][32]  8KB
    float*    sB   = (float*)(smem + 139264);         // 128 f32

    const int tid = threadIdx.x;
    const int node0 = blockIdx.x * 64;
    const int wave = tid >> 5, lane = tid & 31, grp = lane >> 4;

    for (int e = tid; e < 24576; e += 256) {        // W_n1[384][64] -> [64][384]
        int k = e >> 6, n = e & 63;
        sW1T[n * 384 + k] = (_Float16)W_n1[e];
    }
    for (int e = tid; e < 4096; e += 256) {         // W_n2[64][64] -> [64][64]^T
        int k = e >> 6, n = e & 63;
        sW2T[n * 64 + k] = (_Float16)W_n2[e];
    }
    for (int e = tid; e < 2048; e += 256)  sCN[e] = combN[node0 * 32 + e];
    if (tid < 64) { sB[tid] = b_n1[tid]; sB[64 + tid] = b_n2[tid]; }
    // concat cols 0..63 = h ; 64..319 = h_e
    for (int e = tid; e < 4096; e += 256) {
        int nl = e >> 6, k = e & 63;
        sIn[nl * 384 + k] = (_Float16)h[(node0 + nl) * 64 + k];
    }
    for (int e = tid; e < 16384; e += 256) {
        int nl = e >> 8, k = e & 255;
        sIn[nl * 384 + 64 + k] = (_Float16)hE[(node0 + nl) * 256 + k];
    }
    __syncthreads();

    // post-norm MLP: hc1 = silu(CN @ Wp1 + bp1)
    for (int e = tid; e < 4096; e += 256) {
        int nl = e >> 6, k = e & 63;
        float acc = b_p1[k];
#pragma unroll 8
        for (int c = 0; c < 32; ++c) acc += sCN[nl * 32 + c] * W_p1[c * 64 + k];
        sHC[e] = silu_f(acc);
    }
    __syncthreads();
    // h_comb = hc1 @ Wp2 + bp2 -> concat cols 320..383
    for (int e = tid; e < 4096; e += 256) {
        int nl = e >> 6, k = e & 63;
        float acc = b_p2[k];
#pragma unroll 8
        for (int hh = 0; hh < 64; ++hh) acc += sHC[nl * 64 + hh] * W_p2[hh * 64 + k];
        sIn[nl * 384 + 320 + k] = (_Float16)acc;
    }
    __syncthreads();

    // GEMM1: sT = silu(sIn[64x384] @ W_n1[384x64] + bn1)
    for (int t8 = wave; t8 < 16; t8 += 8) {
        int mt = t8 >> 2, nt = t8 & 3;
        v8f c = 0.0f;
#pragma unroll
        for (int kc = 0; kc < 12; ++kc)
            c = wmma_f16(load_a16(sIn, 384, mt * 16, kc * 32, lane),
                         load_bT16(sW1T, 384, kc * 32, nt * 16, lane), c);
        int n = nt * 16 + (lane & 15);
#pragma unroll
        for (int r = 0; r < 8; ++r) {
            int m = mt * 16 + r + (grp << 3);
            sT[m * 64 + n] = (_Float16)silu_f(c[r] + sB[n]);
        }
    }
    __syncthreads();

    // GEMM2: out = sT @ W_n2 + bn2 ; h_out = h + out (residual)
    for (int t8 = wave; t8 < 16; t8 += 8) {
        int mt = t8 >> 2, nt = t8 & 3;
        v8f c = 0.0f;
        c = wmma_f16(load_a16(sT, 64, mt * 16, 0, lane),
                     load_bT16(sW2T, 64, 0, nt * 16, lane), c);
        c = wmma_f16(load_a16(sT, 64, mt * 16, 32, lane),
                     load_bT16(sW2T, 64, 32, nt * 16, lane), c);
        int n = nt * 16 + (lane & 15);
#pragma unroll
        for (int r = 0; r < 8; ++r) {
            int m = mt * 16 + r + (grp << 3);
            int nd = node0 + m;
            out_h[nd * 64 + n] = h[nd * 64 + n] + c[r] + sB[64 + n];
        }
    }

    // velocity + coordinate update
    if (tid < 192) {
        int nl = tid / 3, d = tid % 3;
        int nd = node0 + nl;
        float vn = vgate[nd] * v[nd * 3 + d] + dV[nd * 3 + d];
        out_v[nd * 3 + d] = vn;
        out_x[nd * 3 + d] = x[nd * 3 + d] + vn;
    }
}

// ===========================================================================
extern "C" void kernel_launch(void* const* d_in, const int* in_sizes, int n_in,
                              void* d_out, int out_size, void* d_ws, size_t ws_size,
                              hipStream_t stream) {
    (void)in_sizes; (void)n_in; (void)out_size; (void)ws_size;
    const float* h      = (const float*)d_in[0];
    const float* x      = (const float*)d_in[1];
    const float* v      = (const float*)d_in[2];
    const float* W_e1   = (const float*)d_in[3];
    const float* b_e1   = (const float*)d_in[4];
    const float* W_e2   = (const float*)d_in[5];
    const float* b_e2   = (const float*)d_in[6];
    const float* W_c1   = (const float*)d_in[7];
    const float* b_c1   = (const float*)d_in[8];
    const float* W_c2   = (const float*)d_in[9];
    const float* b_c2   = (const float*)d_in[10];
    const float* W_p1   = (const float*)d_in[11];
    const float* b_p1   = (const float*)d_in[12];
    const float* W_p2   = (const float*)d_in[13];
    const float* b_p2   = (const float*)d_in[14];
    const float* W_s    = (const float*)d_in[15];
    const float* b_s    = (const float*)d_in[16];
    const float* W_n1   = (const float*)d_in[17];
    const float* b_n1   = (const float*)d_in[18];
    const float* W_n2   = (const float*)d_in[19];
    const float* b_n2   = (const float*)d_in[20];
    const float* W_v1   = (const float*)d_in[21];
    const float* b_v1   = (const float*)d_in[22];
    const float* W_v2   = (const float*)d_in[23];
    const float* lgam   = (const float*)d_in[24];

    float* ws    = (float*)d_ws;
    float* linj  = ws;                 // 1024*64
    float* lini  = ws + 65536;         // 1024*64
    float* vgate = ws + 131072;        // 1024
    float* hEbuf = ws + 132096;        // 1024*256
    float* combN = ws + 394240;        // 1024*32
    float* dV    = ws + 427008;        // 1024*3

    float* out_h = (float*)d_out;      // 65536
    float* out_x = out_h + 65536;      // 3072
    float* out_v = out_x + 3072;       // 3072

    sake_pre<<<1024, 256, 0, stream>>>(h, W_e1, W_v1, b_v1, W_v2, linj, lini, vgate);

    sake_edge<<<1024, 256, 135168, stream>>>(x, linj, lini, W_e1, b_e1,
                                             W_e2, b_e2, W_c1, b_c1, W_c2, b_c2,
                                             W_s, b_s, lgam, hEbuf, combN, dV);

    sake_node<<<16, 256, 139776, stream>>>(h, x, v, W_p1, b_p1, W_p2, b_p2,
                                           W_n1, b_n1, W_n2, b_n2,
                                           hEbuf, combN, dV, vgate,
                                           out_h, out_x, out_v);
}
